// MultiHeadAttentionBlock_17892833755758
// MI455X (gfx1250) — matrix operand order repaired and checked by reading, not compiled, not run on hardware
//
#include <hip/hip_runtime.h>
#include <hip/hip_bf16.h>

#define D_MODEL 768
#define NH      12
#define DKH     64
#define NB      2
#define SS      2048
#define MTOT    (NB * SS)   // 4096

typedef __attribute__((ext_vector_type(16))) __bf16 v16bf;
typedef __attribute__((ext_vector_type(8)))  __bf16 v8bf;
typedef __attribute__((ext_vector_type(2)))  __bf16 v2bf;
typedef __attribute__((ext_vector_type(8)))  float  v8f;

#define V8F_ZERO {0.f,0.f,0.f,0.f,0.f,0.f,0.f,0.f}

// WMMA D = A(16x32 bf16) * B(32x16 bf16) + C(16x16 f32)
__device__ __forceinline__ v8f wmma_bf16(v16bf a, v16bf b, v8f c) {
  return __builtin_amdgcn_wmma_f32_16x16x32_bf16(
      /*neg_a=*/false, a, /*neg_b=*/false, b,
      /*c_mod=*/(short)0, c, /*reuse_a=*/false, /*reuse_b=*/false);
}

// A-frag (16-bit, 16xK): lane holds row m=lane&15; element i maps to
// k = 16*(lane>=16) + i  -> 16 contiguous elems.
__device__ __forceinline__ v16bf load_a_frag(const __bf16* rowptr, int lane) {
  const int base = (lane >> 4) << 4;             // 0 or 16
  v8bf lo = *(const v8bf*)(rowptr + base);       // k = base .. base+7
  v8bf hi = *(const v8bf*)(rowptr + base + 8);   // k = base+8 .. base+15
  return __builtin_shufflevector(lo, hi, 0,1,2,3,4,5,6,7,8,9,10,11,12,13,14,15);
}

// ---- CDNA5 async global->LDS copy (ASYNCcnt-tracked), 16 bytes per lane ----
// Flat pointers to LDS carry the LDS byte offset in their low 32 bits
// (aperture check is on addr[63:32]), so truncation yields the DS address.
__device__ __forceinline__ void async_cp16(__bf16* lds_dst, const __bf16* gsrc) {
  uint32_t l = (uint32_t)(uintptr_t)lds_dst;
  unsigned long long g = (unsigned long long)(uintptr_t)gsrc;
  asm volatile("global_load_async_to_lds_b128 %0, %1, off"
               :: "v"(l), "v"(g) : "memory");
}
__device__ __forceinline__ void wait_async_le16() {
  asm volatile("s_wait_asynccnt 0x10" ::: "memory");
}
__device__ __forceinline__ void wait_async_0() {
  asm volatile("s_wait_asynccnt 0x0" ::: "memory");
}

// contiguous 4KB (K chunk: 32 rows x 64 bf16): 8 async b128 per lane
__device__ __forceinline__ void copy_K_chunk(__bf16* dst, const __bf16* src, int lane) {
  #pragma unroll
  for (int j = 0; j < 8; ++j) {
    int off = (j * 32 + lane) * 8;               // 16B units -> bf16 elems
    async_cp16(dst + off, src + off);
  }
}
// V chunk: 64 rows x 32 bf16 (64B) from global rows of stride SS
__device__ __forceinline__ void copy_V_chunk(__bf16* dst, const __bf16* src, int lane) {
  #pragma unroll
  for (int j = 0; j < 8; ++j) {
    int u = j * 32 + lane;                       // 0..255 16B units
    int row = u >> 2, e = (u & 3) << 3;
    async_cp16(dst + row * 32 + e, src + (size_t)row * SS + e);
  }
}

// ---------------------------------------------------------------------------
// Kernel 1: fused QKV projections.  out = x @ W^T + b, stored bf16.
//   grid = (MTOT/16, D_MODEL/128, 3), block = 256 (8 waves, one 16x16 tile each)
//   Q,K stored [B,H,S,64]; V stored transposed [B,H,64,S].
// ---------------------------------------------------------------------------
__global__ __launch_bounds__(256) void qkv_proj_kernel(
    const float* __restrict__ xq, const float* __restrict__ xk, const float* __restrict__ xv,
    const float* __restrict__ Wq, const float* __restrict__ Wk, const float* __restrict__ Wv,
    const float* __restrict__ bq, const float* __restrict__ bk, const float* __restrict__ bv,
    __bf16* __restrict__ qb, __bf16* __restrict__ kb, __bf16* __restrict__ vb)
{
  __shared__ __attribute__((aligned(32))) __bf16 As[16 * 32];

  const int z = blockIdx.z;
  const float* __restrict__ src  = (z == 0) ? xq : (z == 1) ? xk : xv;
  const float* __restrict__ W    = (z == 0) ? Wq : (z == 1) ? Wk : Wv;
  const float* __restrict__ bias = (z == 0) ? bq : (z == 1) ? bk : bv;

  const int m0   = blockIdx.x * 16;
  const int wid  = threadIdx.x >> 5;
  const int lane = threadIdx.x & 31;
  const int nn   = blockIdx.y * 128 + wid * 16 + (lane & 15);
  const int khalf = (lane >> 4) << 4;           // 0 or 16 (B-frag k half)

  // static per-thread staging slot: one float2 -> one packed bf16 pair
  const int t2 = threadIdx.x * 2;               // 0..510
  const int smm = t2 >> 5, skk = t2 & 31;
  const float* sptr = src + (size_t)(m0 + smm) * D_MODEL + skk;

  v8f acc = V8F_ZERO;

  for (int k0 = 0; k0 < D_MODEL; k0 += 32) {
    // stage A 16x32 tile, fp32 -> bf16 (one vectorized load/store per thread)
    {
      float2 f2 = *(const float2*)(sptr + k0);
      v2bf pk; pk[0] = (__bf16)f2.x; pk[1] = (__bf16)f2.y;
      *(v2bf*)(&As[t2]) = pk;
    }
    __syncthreads();

    v16bf afrag = load_a_frag(&As[(lane & 15) * 32], lane);

    // B[k][n] = W[n][k]; lane col n=nn, k = k0+khalf+i (16 contiguous fp32)
    v16bf bfrag;
    const float4* wr = (const float4*)(W + (size_t)nn * D_MODEL + k0 + khalf);
    #pragma unroll
    for (int q4 = 0; q4 < 4; ++q4) {
      float4 w4 = wr[q4];
      bfrag[4*q4+0] = (__bf16)w4.x;
      bfrag[4*q4+1] = (__bf16)w4.y;
      bfrag[4*q4+2] = (__bf16)w4.z;
      bfrag[4*q4+3] = (__bf16)w4.w;
    }

    acc = wmma_bf16(afrag, bfrag, acc);
    __syncthreads();
  }

  // epilogue: C layout -> (m = r + 8*(lane>=16), n = nn)
  const float bval = bias[nn];
  const int h = nn >> 6, d = nn & 63;
  #pragma unroll
  for (int r = 0; r < 8; ++r) {
    int m = m0 + r + ((lane >> 4) << 3);
    int b = m >> 11, s = m & (SS - 1);
    __bf16 val = (__bf16)(acc[r] + bval);
    if (z == 2) {
      vb[(((size_t)b * NH + h) * DKH + d) * SS + s] = val;   // V transposed
    } else {
      __bf16* dst = (z == 0) ? qb : kb;
      dst[(((size_t)b * NH + h) * SS + s) * DKH + d] = val;
    }
  }
}

// ---------------------------------------------------------------------------
// Kernel 2: flash attention, one (b,h,16-row Q tile) per 128-thread block.
//   4 waves split the 2048 kv positions in 32-wide chunks (split-K flash).
//   K/V chunks are double-buffered in LDS via async global->LDS copies
//   (ASYNCcnt): copy chunk j+1 while WMMAs consume chunk j.
// ---------------------------------------------------------------------------
__global__ __launch_bounds__(128) void flash_attn_kernel(
    const __bf16* __restrict__ qb, const __bf16* __restrict__ kb,
    const __bf16* __restrict__ vb, const int* __restrict__ mask,
    __bf16* __restrict__ ob)
{
  __shared__ __attribute__((aligned(32))) __bf16 Kbuf[4][2][32 * DKH]; // 32 KB
  __shared__ __attribute__((aligned(32))) __bf16 Vbuf[4][2][DKH * 32]; // 32 KB
  __shared__ __attribute__((aligned(32))) __bf16 Pst[4][16 * 32];      //  4 KB
  __shared__ __attribute__((aligned(16))) float  accS[4][16 * DKH];    // 16 KB
  __shared__ float mS[4][16], lS[4][16];

  const int qt = blockIdx.x, h = blockIdx.y, b = blockIdx.z;
  const int wid  = threadIdx.x >> 5;
  const int lane = threadIdx.x & 31;
  const int nsub  = lane & 15;
  const int gsel  = lane >> 4;                  // half-wave (0/1)
  const int khalf = gsel << 4;

  const __bf16* Qp = qb + (((size_t)b * NH + h) * SS + qt * 16) * DKH;
  const __bf16* Kp = kb + ((size_t)b * NH + h) * SS * DKH;
  const __bf16* Vp = vb + ((size_t)b * NH + h) * DKH * SS;   // [64][2048]
  const int*    Mp = mask + (size_t)b * SS * SS + (size_t)(qt * 16) * SS;

  // Q A-frags for d=0..31 and d=32..63 (resident for the whole kernel)
  v16bf qa0 = load_a_frag(Qp + (size_t)(lane & 15) * DKH,      lane);
  v16bf qa1 = load_a_frag(Qp + (size_t)(lane & 15) * DKH + 32, lane);

  float m_run[8], l_run[8];
  #pragma unroll
  for (int r = 0; r < 8; ++r) { m_run[r] = -3.0e38f; l_run[r] = 0.f; }
  v8f o0 = V8F_ZERO, o1 = V8F_ZERO, o2 = V8F_ZERO, o3 = V8F_ZERO;

  // prime the pipeline: async-copy this wave's first chunk
  {
    const int k0 = wid * 32;
    copy_K_chunk(&Kbuf[wid][0][0], Kp + (size_t)k0 * DKH, lane);
    copy_V_chunk(&Vbuf[wid][0][0], Vp + k0, lane);
  }

  for (int j = 0; j < 16; ++j) {                // uniform across all 4 waves
    const int cur = j & 1;
    const int k0  = (wid + 4 * j) * 32;

    if (j + 1 < 16) {
      const int kn = (wid + 4 * (j + 1)) * 32;
      copy_K_chunk(&Kbuf[wid][cur ^ 1][0], Kp + (size_t)kn * DKH, lane);
      copy_V_chunk(&Vbuf[wid][cur ^ 1][0], Vp + kn, lane);
      wait_async_le16();   // 16 newest still in flight; current chunk resident
    } else {
      wait_async_0();
    }

    const __bf16* Kl = &Kbuf[wid][cur][0];      // [32][64]
    const __bf16* Vl = &Vbuf[wid][cur][0];      // [64][32]

    // ---- scores S = Q * K^T : two 16x16 f32 tiles over cols [k0, k0+32)
    v8f s0 = V8F_ZERO, s1 = V8F_ZERO;
    {
      const __bf16* krow0 = Kl + (size_t)nsub * DKH + khalf;
      v16bf kf0a = *(const v16bf*)(krow0);        // d = khalf..+15
      v16bf kf0b = *(const v16bf*)(krow0 + 32);   // d = 32+khalf..+15
      s0 = wmma_bf16(qa0, kf0a, s0);
      s0 = wmma_bf16(qa1, kf0b, s0);
      const __bf16* krow1 = Kl + (size_t)(16 + nsub) * DKH + khalf;
      v16bf kf1a = *(const v16bf*)(krow1);
      v16bf kf1b = *(const v16bf*)(krow1 + 32);
      s1 = wmma_bf16(qa0, kf1a, s1);
      s1 = wmma_bf16(qa1, kf1b, s1);
    }

    // ---- mask + online softmax (row stats reduced across 16-lane N group)
    #pragma unroll
    for (int r = 0; r < 8; ++r) {
      const int row = r + (gsel << 3);
      const int* mrow = Mp + (size_t)row * SS + k0 + nsub;
      float x0 = (mrow[0]  == 0) ? -1e9f : s0[r] * 0.125f;   // 1/sqrt(64)
      float x1 = (mrow[16] == 0) ? -1e9f : s1[r] * 0.125f;
      float mx = fmaxf(x0, x1);
      #pragma unroll
      for (int off = 1; off < 16; off <<= 1)
        mx = fmaxf(mx, __shfl_xor(mx, off, 32));
      float mnew = fmaxf(m_run[r], mx);
      float p0 = __expf(x0 - mnew);
      float p1 = __expf(x1 - mnew);
      float rs = __expf(m_run[r] - mnew);
      float ps = p0 + p1;
      #pragma unroll
      for (int off = 1; off < 16; off <<= 1)
        ps += __shfl_xor(ps, off, 32);
      l_run[r] = l_run[r] * rs + ps;
      m_run[r] = mnew;
      o0[r] = o0[r] * rs; o1[r] = o1[r] * rs;
      o2[r] = o2[r] * rs; o3[r] = o3[r] * rs;
      // P (C layout) -> LDS row-major 16x32 for A-layout reload
      Pst[wid][row * 32 + nsub]      = (__bf16)p0;
      Pst[wid][row * 32 + nsub + 16] = (__bf16)p1;
    }
    __syncthreads();

    // ---- O += P(16x32) * V(32x64): 4 WMMAs (one per 16-wide d tile)
    v16bf pa = load_a_frag(&Pst[wid][(lane & 15) * 32], lane);
    #pragma unroll
    for (int t = 0; t < 4; ++t) {
      const __bf16* vrow = Vl + (size_t)(t * 16 + nsub) * 32 + khalf;
      v16bf vf = *(const v16bf*)(vrow);           // kv = khalf..+15
      if      (t == 0) o0 = wmma_bf16(pa, vf, o0);
      else if (t == 1) o1 = wmma_bf16(pa, vf, o1);
      else if (t == 2) o2 = wmma_bf16(pa, vf, o2);
      else             o3 = wmma_bf16(pa, vf, o3);
    }
    __syncthreads();
  }

  // ---- merge the 4 per-wave partials
  #pragma unroll
  for (int r = 0; r < 8; ++r) {
    const int row = r + (gsel << 3);
    accS[wid][row * DKH + nsub +  0] = o0[r];
    accS[wid][row * DKH + nsub + 16] = o1[r];
    accS[wid][row * DKH + nsub + 32] = o2[r];
    accS[wid][row * DKH + nsub + 48] = o3[r];
  }
  if (nsub == 0) {
    #pragma unroll
    for (int r = 0; r < 8; ++r) {
      const int row = r + (gsel << 3);
      mS[wid][row] = m_run[r];
      lS[wid][row] = l_run[r];
    }
  }
  __syncthreads();

  for (int t = threadIdx.x; t < 16 * DKH; t += 128) {
    const int row = t >> 6, d = t & 63;
    float mmax = fmaxf(fmaxf(mS[0][row], mS[1][row]),
                       fmaxf(mS[2][row], mS[3][row]));
    float ltot = 0.f, osum = 0.f;
    #pragma unroll
    for (int w = 0; w < 4; ++w) {
      float sc = __expf(mS[w][row] - mmax);
      ltot += sc * lS[w][row];
      osum += sc * accS[w][t];
    }
    const int s = qt * 16 + row;
    ob[(size_t)(b * SS + s) * D_MODEL + h * DKH + d] = (__bf16)(osum / ltot);
  }
}

// ---------------------------------------------------------------------------
// Kernel 3: output projection  out = O @ Wo^T + bo  (fp32 out)
//   grid = (MTOT/16, D_MODEL/128), block = 256 (8 waves)
// ---------------------------------------------------------------------------
__global__ __launch_bounds__(256) void out_proj_kernel(
    const __bf16* __restrict__ A, const float* __restrict__ Wo,
    const float* __restrict__ bo, float* __restrict__ out)
{
  const int m0   = blockIdx.x * 16;
  const int wid  = threadIdx.x >> 5;
  const int lane = threadIdx.x & 31;
  const int nn   = blockIdx.y * 128 + wid * 16 + (lane & 15);
  const int khalf = (lane >> 4) << 4;

  v8f acc = V8F_ZERO;
  const __bf16* arow = A + (size_t)(m0 + (lane & 15)) * D_MODEL;

  for (int k0 = 0; k0 < D_MODEL; k0 += 32) {
    v16bf afrag = load_a_frag(arow + k0, lane);
    v16bf bfrag;
    const float4* wr = (const float4*)(Wo + (size_t)nn * D_MODEL + k0 + khalf);
    #pragma unroll
    for (int q4 = 0; q4 < 4; ++q4) {
      float4 w4 = wr[q4];
      bfrag[4*q4+0] = (__bf16)w4.x;
      bfrag[4*q4+1] = (__bf16)w4.y;
      bfrag[4*q4+2] = (__bf16)w4.z;
      bfrag[4*q4+3] = (__bf16)w4.w;
    }
    acc = wmma_bf16(afrag, bfrag, acc);
  }

  const float bval = bo[nn];
  #pragma unroll
  for (int r = 0; r < 8; ++r) {
    int m = m0 + r + ((lane >> 4) << 3);
    out[(size_t)m * D_MODEL + nn] = acc[r] + bval;
  }
}

// ---------------------------------------------------------------------------
extern "C" void kernel_launch(void* const* d_in, const int* in_sizes, int n_in,
                              void* d_out, int out_size, void* d_ws, size_t ws_size,
                              hipStream_t stream) {
  const float* q    = (const float*)d_in[0];
  const float* k    = (const float*)d_in[1];
  const float* v    = (const float*)d_in[2];
  const int*   mask = (const int*)  d_in[3];
  const float* Wq   = (const float*)d_in[4];
  const float* bq   = (const float*)d_in[5];
  const float* Wk   = (const float*)d_in[6];
  const float* bk   = (const float*)d_in[7];
  const float* Wv   = (const float*)d_in[8];
  const float* bv   = (const float*)d_in[9];
  const float* Wo   = (const float*)d_in[10];
  const float* bo   = (const float*)d_in[11];
  float* out = (float*)d_out;

  // workspace: bf16 Q,K,V,O buffers (4 * 4096*768*2B = 24 MiB)
  const size_t mat = (size_t)MTOT * D_MODEL;
  char* ws = (char*)d_ws;
  __bf16* qb = (__bf16*)(ws);
  __bf16* kb = (__bf16*)(ws + 2 * mat);
  __bf16* vb = (__bf16*)(ws + 4 * mat);
  __bf16* ob = (__bf16*)(ws + 6 * mat);

  dim3 g1(MTOT / 16, D_MODEL / 128, 3);
  qkv_proj_kernel<<<g1, 256, 0, stream>>>(q, k, v, Wq, Wk, Wv, bq, bk, bv,
                                          qb, kb, vb);

  dim3 g2(SS / 16, NH, NB);
  flash_attn_kernel<<<g2, 128, 0, stream>>>(qb, kb, vb, mask, ob);

  dim3 g3(MTOT / 16, D_MODEL / 128);
  out_proj_kernel<<<g3, 256, 0, stream>>>(ob, Wo, bo, out);
}
